// VectorQuantizer_37512244363340
// MI455X (gfx1250) — matrix operand order repaired and checked by reading, not compiled, not run on hardware
//
#include <hip/hip_runtime.h>

// ---------------------------------------------------------------------------
// VQ nearest-code search for MI455X (gfx1250, wave32, WMMA + TDM).
// dists row-constant ||x||^2 dropped:  val_k = ||e_k||^2 - 2 * <x, e_k>
// <x,e> computed with split-precision bf16 WMMA:
//   x ~= xh + xl, e ~= eh + el;  <x,e> ~= xh.eh + xh.el + xl.eh  (f32 accum)
// z_e tile staged to LDS via the Tensor Data Mover when available.
// ---------------------------------------------------------------------------

typedef __attribute__((ext_vector_type(16))) __bf16        bf16x16;
typedef __attribute__((ext_vector_type(8)))  float         floatx8;
typedef __attribute__((ext_vector_type(8)))  unsigned int  uintx8;
typedef __attribute__((ext_vector_type(4)))  unsigned int  uint32x4;
typedef __attribute__((ext_vector_type(8)))  int           int32x8;
typedef __attribute__((ext_vector_type(4)))  int           int32x4;

#define NB    32
#define CDIM  256
#define HH    32
#define WW    32
#define KCODE 1024
#define ZQ_ELEMS (NB * CDIM * HH * WW)   // 8388608

#if __has_builtin(__builtin_amdgcn_tensor_load_to_lds) && \
    __has_builtin(__builtin_amdgcn_s_wait_tensorcnt)
#define USE_TDM 1
#else
#define USE_TDM 0
#endif

__device__ __forceinline__ unsigned short f32_to_bf16(float x) {
  unsigned u = __builtin_bit_cast(unsigned, x);
  unsigned r = u + 0x7FFFu + ((u >> 16) & 1u);   // round-to-nearest-even
  return (unsigned short)(r >> 16);
}
__device__ __forceinline__ float bf16_to_f32(unsigned short b) {
  return __builtin_bit_cast(float, (unsigned)b << 16);
}
// pack two fp32 into one dword of bf16-hi pair; residual lo pair via out-param
__device__ __forceinline__ unsigned split_pack(float x0, float x1, unsigned &lo) {
  unsigned short h0 = f32_to_bf16(x0), h1 = f32_to_bf16(x1);
  unsigned short l0 = f32_to_bf16(x0 - bf16_to_f32(h0));
  unsigned short l1 = f32_to_bf16(x1 - bf16_to_f32(h1));
  lo = (unsigned)l0 | ((unsigned)l1 << 16);
  return (unsigned)h0 | ((unsigned)h1 << 16);
}

__device__ __forceinline__ floatx8 wmma_bf16(uintx8 a, uintx8 b, floatx8 c) {
  return __builtin_amdgcn_wmma_f32_16x16x32_bf16(
      false, __builtin_bit_cast(bf16x16, a),
      false, __builtin_bit_cast(bf16x16, b),
      (short)0, c, false, false);
}

// --------------------------- kernel 1a: code norms -------------------------
__global__ void vq_norms(const float* __restrict__ cb, float* __restrict__ cnorm) {
  int k = blockIdx.x * blockDim.x + threadIdx.x;
  if (k < KCODE) {
    const float* r = cb + k * CDIM;
    float s = 0.f;
#pragma unroll 4
    for (int c = 0; c < CDIM; ++c) s += r[c] * r[c];
    cnorm[k] = s;
  }
}

// ----------------- kernel 1b: codebook -> WMMA-B fragment layout -----------
// B as packed here (32x16 bf16, contiguous k, same order as A): lanes 0-15: col N=L, K=0..15 (V_v holds K=2v,2v+1)
//                          lanes 16-31: col N=L-16, K=16..31
// dword index g = ((nt*8 + k8)*32 + L)*8 + v,  nt in [0,64), k8 in [0,8)
__global__ void vq_pack_codebook(const float* __restrict__ cb,
                                 unsigned* __restrict__ bHi,
                                 unsigned* __restrict__ bLo) {
  unsigned g  = blockIdx.x * 256u + threadIdx.x;     // 512 blocks -> 131072
  unsigned v  = g & 7u;
  unsigned L  = (g >> 3) & 31u;
  unsigned k8 = (g >> 8) & 7u;
  unsigned nt = g >> 11;
  unsigned n  = nt * 16u + (L & 15u);
  unsigned c0 = k8 * 32u + ((L >> 4) * 16u) + 2u * v;
  float x0 = cb[n * CDIM + c0];
  float x1 = cb[n * CDIM + c0 + 1u];
  unsigned lo;
  unsigned hi = split_pack(x0, x1, lo);
  bHi[g] = hi;
  bLo[g] = lo;
}

// ------------------------------ kernel 2: main -----------------------------
// One block per (b,h): rows = 32 consecutive w. 256 threads = 8 wave32.
// Each wave handles 128 codes (8 N-tiles); M = 32 rows = 2 M-tiles.
__global__ __launch_bounds__(256) void vq_main(
    const float* __restrict__ ze, const float* __restrict__ cb,
    const unsigned* __restrict__ bHi, const unsigned* __restrict__ bLo,
    const float* __restrict__ cnorm,
    float* __restrict__ outq, float* __restrict__ outidx) {
  __shared__ alignas(16) float zf[CDIM * WW];        // 32 KB fp32 tile [c][w]
  __shared__ alignas(32) unsigned aHi[2 * 8 * 32 * 8];  // 16 KB A-frag hi
  __shared__ alignas(32) unsigned aLo[2 * 8 * 32 * 8];  // 16 KB A-frag lo
  __shared__ float redv[8 * 32];
  __shared__ int   redi[8 * 32];
  __shared__ int   fidx[32];

  const int t  = threadIdx.x;
  const int bh = blockIdx.x;                 // b*32 + h
  const int b  = bh >> 5, h = bh & 31;
  const float* zeb = ze + ((size_t)b * CDIM * (HH * WW) + h * WW);

  // ---- phase 1: stage z_e tile (256 c x 32 w, row stride 1024) into LDS ----
#if USE_TDM
  if (t < 32) {   // one wave issues the DMA; TDM ignores EXEC
    unsigned long long ga = (unsigned long long)(uintptr_t)zeb;
    unsigned lds_off = (unsigned)(uintptr_t)&zf[0];  // low 32b of LDS aperture
    uint32x4 g0;
    g0[0] = 1u;                                   // count=1 valid descriptor
    g0[1] = lds_off;                              // D#.lds_addr
    g0[2] = (unsigned)ga;                         // global_addr[31:0]
    g0[3] = ((unsigned)(ga >> 32) & 0x01FFFFFFu)  // global_addr[56:32]
            | 0x80000000u;                        // type=2 ("image")
    int32x8 g1;
    g1[0] = 0x00020000;          // wg_mask=0, data_size=2 (4B), no pad/iter
    g1[1] = (int)(32u << 16);    // tensor_dim0[15:0]=32 in [31:16]
    g1[2] = (int)(256u << 16);   // tensor_dim0[31:16]=0, tensor_dim1[15:0]=256
    g1[3] = (int)(32u << 16);    // tensor_dim1[31:16]=0, tile_dim0=32
    g1[4] = 256;                 // tile_dim1=256, tile_dim2=0
    g1[5] = 1024;                // tensor_dim0_stride[31:0] = 1024 elements
    g1[6] = 0;                   // stride hi, tensor_dim1_stride lo
    g1[7] = 0;
    int32x4 gz4 = {0, 0, 0, 0};            // groups 2/3 unused (2D tensor)
    int32x8 gz8 = {0, 0, 0, 0, 0, 0, 0, 0};
    __builtin_amdgcn_tensor_load_to_lds(g0, g1, gz4, gz4, gz8, 0);
    __builtin_amdgcn_s_wait_tensorcnt(0);
  }
#else
#pragma unroll 4
  for (int i = 0; i < 32; ++i) {
    int idx = i * 256 + t;
    int c = idx >> 5, w = idx & 31;
    zf[c * 32 + w] = zeb[(size_t)c * (HH * WW) + w];
  }
#endif
  __syncthreads();

  // ---- phase 2: split fp32 -> bf16 hi/lo, packed in WMMA A-fragment layout
  // A packed here (16x32 bf16, contiguous k, same order as B): lanes 0-15 row M=L: V0..7 K=0..15
  //                 lanes 16-31 row M=L-16: V0..7 K=16..31
  for (int i = 0; i < 16; ++i) {
    int g  = i * 256 + t;
    int v  = g & 7, L = (g >> 3) & 31, k8 = (g >> 8) & 7, m = g >> 11;
    int kb = (L >> 4) * 8;
    int c0 = k8 * 32 + 2 * kb + 2 * v;
    int w  = m * 16 + (L & 15);
    float x0 = zf[c0 * 32 + w];
    float x1 = zf[(c0 + 1) * 32 + w];
    unsigned lo;
    unsigned hi = split_pack(x0, x1, lo);
    aHi[g] = hi;
    aLo[g] = lo;
  }
  __syncthreads();

  // ---- GEMM + running argmin -------------------------------------------
  const int wv = t >> 5, lane = t & 31;
  float mv0[8], mv1[8];
  int   mi0[8], mi1[8];
#pragma unroll
  for (int r = 0; r < 8; ++r) {
    mv0[r] = 3.402823466e+38f; mv1[r] = 3.402823466e+38f;
    mi0[r] = 0;                mi1[r] = 0;
  }

  for (int nn = 0; nn < 8; ++nn) {
    const int nt = wv * 8 + nn;
    __builtin_prefetch(&bHi[((size_t)(nt * 8) * 32 + lane) * 8], 0, 3);
    floatx8 acc0 = {0.f, 0.f, 0.f, 0.f, 0.f, 0.f, 0.f, 0.f};
    floatx8 acc1 = {0.f, 0.f, 0.f, 0.f, 0.f, 0.f, 0.f, 0.f};
    for (int k = 0; k < 8; ++k) {
      const uintx8 a0h = *(const uintx8*)&aHi[((0 + k) * 32 + lane) * 8];
      const uintx8 a0l = *(const uintx8*)&aLo[((0 + k) * 32 + lane) * 8];
      const uintx8 a1h = *(const uintx8*)&aHi[((8 + k) * 32 + lane) * 8];
      const uintx8 a1l = *(const uintx8*)&aLo[((8 + k) * 32 + lane) * 8];
      const size_t boff = ((size_t)(nt * 8 + k) * 32 + lane) * 8;
      const uintx8 bh8 = *(const uintx8*)&bHi[boff];
      const uintx8 bl8 = *(const uintx8*)&bLo[boff];
      acc0 = wmma_bf16(a0h, bh8, acc0);
      acc0 = wmma_bf16(a0h, bl8, acc0);
      acc0 = wmma_bf16(a0l, bh8, acc0);
      acc1 = wmma_bf16(a1h, bh8, acc1);
      acc1 = wmma_bf16(a1h, bl8, acc1);
      acc1 = wmma_bf16(a1l, bh8, acc1);
    }
    const int   code = nt * 16 + (lane & 15);
    const float cn   = cnorm[code];
#pragma unroll
    for (int r = 0; r < 8; ++r) {
      float v0 = fmaf(-2.f, acc0[r], cn);
      if (v0 < mv0[r]) { mv0[r] = v0; mi0[r] = code; }
      float v1 = fmaf(-2.f, acc1[r], cn);
      if (v1 < mv1[r]) { mv1[r] = v1; mi1[r] = code; }
    }
  }

  // ---- reduce across 16 lanes sharing a row (xor masks stay in half-wave)
#pragma unroll
  for (int r = 0; r < 8; ++r) {
    float v0 = mv0[r]; int i0 = mi0[r];
    float v1 = mv1[r]; int i1 = mi1[r];
#pragma unroll
    for (int off = 1; off < 16; off <<= 1) {
      float ov = __shfl_xor(v0, off, 32); int oi = __shfl_xor(i0, off, 32);
      if (ov < v0 || (ov == v0 && oi < i0)) { v0 = ov; i0 = oi; }
      ov = __shfl_xor(v1, off, 32); oi = __shfl_xor(i1, off, 32);
      if (ov < v1 || (ov == v1 && oi < i1)) { v1 = ov; i1 = oi; }
    }
    if ((lane & 15) == 0) {
      int rb = (lane >> 4) * 8;   // lane 0 -> rows r / 16+r ; lane16 -> +8
      redv[wv * 32 + rb + r]      = v0;  redi[wv * 32 + rb + r]      = i0;
      redv[wv * 32 + 16 + rb + r] = v1;  redi[wv * 32 + 16 + rb + r] = i1;
    }
  }
  __syncthreads();

  // ---- reduce across the 8 waves, emit indices --------------------------
  if (t < 32) {
    float bv = redv[t]; int bi = redi[t];
#pragma unroll
    for (int w8 = 1; w8 < 8; ++w8) {
      float v = redv[w8 * 32 + t]; int ii = redi[w8 * 32 + t];
      if (v < bv || (v == bv && ii < bi)) { bv = v; bi = ii; }
    }
    fidx[t] = bi;
    outidx[bh * 32 + t] = (float)bi;   // idx concatenated after z_q as float
  }
  __syncthreads();

  // ---- gather z_q = codebook[idx] into [B,C,H,W], coalesced stores ------
#pragma unroll 4
  for (int i = 0; i < 32; ++i) {
    int idx = i * 256 + t;
    int c = idx >> 5, w = idx & 31;
    outq[((size_t)b * CDIM + c) * (HH * WW) + h * WW + w] =
        cb[(size_t)fidx[w] * CDIM + c];
  }
}

// ---------------------------------------------------------------------------
extern "C" void kernel_launch(void* const* d_in, const int* in_sizes, int n_in,
                              void* d_out, int out_size, void* d_ws, size_t ws_size,
                              hipStream_t stream) {
  (void)in_sizes; (void)n_in; (void)out_size; (void)ws_size;
  const float* z_e = (const float*)d_in[0];
  const float* cb  = (const float*)d_in[1];

  // workspace: [cnorm 4KB][bHi 512KB][bLo 512KB]
  float*    cnorm = (float*)d_ws;
  unsigned* bHi   = (unsigned*)((char*)d_ws + 4096);
  unsigned* bLo   = bHi + (KCODE * CDIM / 2);      // 131072 dwords each

  float* outq   = (float*)d_out;
  float* outidx = outq + ZQ_ELEMS;

  vq_norms<<<4, 256, 0, stream>>>(cb, cnorm);
  vq_pack_codebook<<<512, 256, 0, stream>>>(cb, bHi, bLo);
  vq_main<<<NB * HH, 256, 0, stream>>>(z_e, cb, bHi, bLo, cnorm, outq, outidx);
}
